// MLPExperts_22643067585119
// MI455X (gfx1250) — compile-verified, matrix operand order checked by reading, not run on hardware
//
#include <hip/hip_runtime.h>
#include <cmath>

// ---------------------------------------------------------------------------
// MI455X (gfx1250) MoE-MLP: out = (gelu(x@W1 + b1)) @ W2 + b2, 8 experts.
// Compute-bound (275 GFLOP vs ~450MB HBM traffic) -> run the GEMMs on the
// bf16 WMMA pipe (v_wmma_f32_16x16x32_bf16) with f32 accumulation.
// Pass 1 writes gelu(h) as bf16 into d_ws (128 MB), pass 2 reads it.
// Pass 2's A tile (already bf16) is copied with async global->LDS (ASYNCcnt).
// Tile loads are gather-then-commit so all global loads are in flight at once.
// ---------------------------------------------------------------------------

typedef __bf16 bf16_t;
typedef bf16_t v16bf __attribute__((ext_vector_type(16)));
typedef bf16_t v8bf  __attribute__((ext_vector_type(8)));
typedef float  v8f   __attribute__((ext_vector_type(8)));

// Types for the async global->LDS builtin (probe: arg0 is AS1 ptr to int[4] vec)
typedef int v4i_b128 __attribute__((vector_size(16)));
typedef __attribute__((address_space(1))) v4i_b128 as1_v4i;
typedef __attribute__((address_space(3))) v4i_b128 as3_v4i;

#define BM 128
#define BN 128
#define BK 32
#define LDT 56   // padded LDS row stride (elements): 112B rows -> 16B aligned frags

#if defined(__gfx1250__) && __has_builtin(__builtin_amdgcn_global_load_async_to_lds_b128) && \
    __has_builtin(__builtin_amdgcn_s_wait_asynccnt)
#define HAVE_ASYNC_LDS 1
#else
#define HAVE_ASYNC_LDS 0
#endif

__device__ __forceinline__ float gelu_tanh_f(float x) {
    const float c = 0.7978845608028654f;
    float t = tanhf(c * (x + 0.044715f * x * x * x));
    return 0.5f * x * (1.0f + t);
}

// Load a 16-element bf16 fragment for one lane: two contiguous 16B chunks
// (ISA 16-bit A/B layout: lanes 0-15 hold K 0-7 & 16-23, lanes 16-31 K 8-15 & 24-31)
__device__ __forceinline__ v16bf load_frag(const bf16_t* p) {
    v8bf lo = *(const v8bf*)(p);
    v8bf hi = *(const v8bf*)(p + 16);
    return __builtin_shufflevector(lo, hi, 0,1,2,3,4,5,6,7,8,9,10,11,12,13,14,15);
}

__device__ __forceinline__ void store_out(bf16_t* p, float v) { *p = (bf16_t)v; }
__device__ __forceinline__ void store_out(float*  p, float v) { *p = v; }

// A_BF16: A operand already bf16 (pass 2). GELU_OUT: apply gelu in epilogue (pass 1).
template <bool A_BF16, bool GELU_OUT, typename OutT>
__global__ __launch_bounds__(256)
void mlp_gemm_kernel(const void* __restrict__ Av,
                     const float* __restrict__ Bw,
                     const float* __restrict__ bias,
                     OutT* __restrict__ C,
                     int M, int N, int K)
{
    __shared__ bf16_t sA[2][BM * LDT];
    __shared__ bf16_t sB[2][BN * LDT];   // stored transposed: sB[n][k]

    const int tid  = threadIdx.x;
    const int lane = tid & 31;
    const int wid  = tid >> 5;           // 8 wave32s
    const int e    = blockIdx.z;
    const int mblk = blockIdx.y * BM;
    const int nblk = blockIdx.x * BN;

    const int wm = (wid & 1) * 64;       // wave tile: 64 (M) x 32 (N)
    const int wn = (wid >> 1) * 32;

    const int hl = lane >> 4;            // half-wave select per ISA layout
    const int lr = lane & 15;

    const float* Bwe = Bw + (size_t)e * K * N;

    v8f acc[4][2];
    #pragma unroll
    for (int mi = 0; mi < 4; ++mi)
        #pragma unroll
        for (int ni = 0; ni < 2; ++ni)
            acc[mi][ni] = v8f{};

    auto load_tiles = [&](int kbase, int buf) {
        // ---------- B tile gather: 32 (K) x 128 (N) fp32 weights ----------
        float4 bv[4];
        #pragma unroll
        for (int i = 0; i < 4; ++i) {
            int q  = tid + i * 256;
            int k  = q >> 5;
            int n0 = (q & 31) * 4;
            bv[i] = *(const float4*)(Bwe + (size_t)(kbase + k) * N + nblk + n0);
        }

        // ---------- A tile: 128 x 32 ----------
        if constexpr (A_BF16) {
            const bf16_t* Ae = (const bf16_t*)Av + (size_t)e * M * K;
            #pragma unroll
            for (int i = 0; i < 2; ++i) {            // 512 x 16B chunks / 256 thr
                int q   = tid + i * 256;
                int row = q >> 2;
                int c8  = (q & 3) * 8;
                const bf16_t* g = Ae + (size_t)(mblk + row) * K + kbase + c8;
                bf16_t*       l = &sA[buf][row * LDT + c8];
#if HAVE_ASYNC_LDS
                // CDNA5 async global->LDS copy: bypasses VGPRs, tracked by ASYNCcnt
                __builtin_amdgcn_global_load_async_to_lds_b128(
                    (as1_v4i*)g, (as3_v4i*)l, /*offset=*/0, /*cpol=*/0);
#else
                *(uint4*)l = *(const uint4*)g;
#endif
            }
        } else {
            const float* Ae = (const float*)Av + (size_t)e * M * K;
            float4 av[4];
            #pragma unroll
            for (int i = 0; i < 4; ++i) {            // gather: 1024 x float4 / 256 thr
                int q   = tid + i * 256;
                int row = q >> 3;
                int c4  = (q & 7) * 4;
                av[i] = *(const float4*)(Ae + (size_t)(mblk + row) * K + kbase + c4);
            }
            #pragma unroll
            for (int i = 0; i < 4; ++i) {            // commit: cvt_pk_bf16 + ds_store
                int q   = tid + i * 256;
                int row = q >> 3;
                int c4  = (q & 7) * 4;
                bf16_t* d = &sA[buf][row * LDT + c4];
                d[0] = (bf16_t)av[i].x; d[1] = (bf16_t)av[i].y;
                d[2] = (bf16_t)av[i].z; d[3] = (bf16_t)av[i].w;
            }
        }

        // ---------- B tile commit (transposed store) ----------
        #pragma unroll
        for (int i = 0; i < 4; ++i) {
            int q  = tid + i * 256;
            int k  = q >> 5;
            int n0 = (q & 31) * 4;
            sB[buf][(n0 + 0) * LDT + k] = (bf16_t)bv[i].x;
            sB[buf][(n0 + 1) * LDT + k] = (bf16_t)bv[i].y;
            sB[buf][(n0 + 2) * LDT + k] = (bf16_t)bv[i].z;
            sB[buf][(n0 + 3) * LDT + k] = (bf16_t)bv[i].w;
        }
    };

    load_tiles(0, 0);
#if HAVE_ASYNC_LDS
    if constexpr (A_BF16) __builtin_amdgcn_s_wait_asynccnt(0);
#endif
    __syncthreads();

    const int nk = K / BK;
    for (int kk = 0; kk < nk; ++kk) {
        const int buf = kk & 1;
        if (kk + 1 < nk) load_tiles((kk + 1) * BK, buf ^ 1);

        v16bf afrag[4], bfrag[2];
        #pragma unroll
        for (int mi = 0; mi < 4; ++mi)
            afrag[mi] = load_frag(&sA[buf][(wm + mi * 16 + lr) * LDT + hl * 8]);
        #pragma unroll
        for (int ni = 0; ni < 2; ++ni)
            bfrag[ni] = load_frag(&sB[buf][(wn + ni * 16 + lr) * LDT + hl * 8]);

        #pragma unroll
        for (int mi = 0; mi < 4; ++mi)
            #pragma unroll
            for (int ni = 0; ni < 2; ++ni)
                acc[mi][ni] = __builtin_amdgcn_wmma_f32_16x16x32_bf16(
                    /*neg_a=*/false, afrag[mi],
                    /*neg_b=*/false, bfrag[ni],
                    /*c_mod=*/(short)0, acc[mi][ni],
                    /*reuse_a=*/false, /*reuse_b=*/false);

#if HAVE_ASYNC_LDS
        if constexpr (A_BF16) __builtin_amdgcn_s_wait_asynccnt(0);
#endif
        __syncthreads();
    }

    // ---- Epilogue: bias (+ optional GELU), C/D layout: VGPR r -> row, lane -> col
    const float* be = bias + (size_t)e * N;
    OutT* Ce = C + (size_t)e * M * N;
    #pragma unroll
    for (int ni = 0; ni < 2; ++ni) {
        int col  = nblk + wn + ni * 16 + lr;
        float bv = be[col];
        #pragma unroll
        for (int mi = 0; mi < 4; ++mi) {
            #pragma unroll
            for (int r = 0; r < 8; ++r) {
                int row = mblk + wm + mi * 16 + hl * 8 + r;
                float v = acc[mi][ni][r] + bv;
                if constexpr (GELU_OUT) v = gelu_tanh_f(v);
                store_out(&Ce[(size_t)row * N + col], v);
            }
        }
    }
}

extern "C" void kernel_launch(void* const* d_in, const int* in_sizes, int n_in,
                              void* d_out, int out_size, void* d_ws, size_t ws_size,
                              hipStream_t stream) {
    const float* x  = (const float*)d_in[0];   // [8, 2048, 1024]
    const float* w1 = (const float*)d_in[1];   // [8, 1024, 4096]
    const float* b1 = (const float*)d_in[2];   // [8, 1, 4096]
    const float* w2 = (const float*)d_in[3];   // [8, 4096, 1024]
    const float* b2 = (const float*)d_in[4];   // [8, 1, 1024]
    float*  out = (float*)d_out;               // [8, 2048, 1024]
    bf16_t* h   = (bf16_t*)d_ws;               // [8, 2048, 4096] bf16 = 128 MB

    const int E = 8, T = 2048, D = 1024, F = 4096;
    dim3 blk(256);

    // Pass 1: h = gelu(x @ W1 + b1)   (M=2048, N=4096, K=1024)
    dim3 g1(F / BN, T / BM, E);
    mlp_gemm_kernel<false, true, bf16_t><<<g1, blk, 0, stream>>>(x, w1, b1, h, T, F, D);

    // Pass 2: out = h @ W2 + b2       (M=2048, N=1024, K=4096)
    dim3 g2(D / BN, T / BM, E);
    mlp_gemm_kernel<true, false, float><<<g2, blk, 0, stream>>>(h, w2, b2, out, T, D, F);
}
